// DSIN_84207128805997
// MI455X (gfx1250) — compile-verified, matrix operand order checked by reading, not run on hardware
//
#include <hip/hip_runtime.h>
#include <hip/hip_bf16.h>
#include <cstdint>
#include <cstddef>

// Problem constants (match reference)
#define Bn 4096
#define Tn 50
#define Dn 128
#define Pn 64
#define Mn (Bn * Tn)   // 204800 rows, divisible by 128

typedef __attribute__((ext_vector_type(16))) _Float16 v16h;
typedef __attribute__((ext_vector_type(8)))  _Float16 v8h;
typedef __attribute__((ext_vector_type(8)))  float    v8f;

#define NEG_INF (-1.0e30f)

// ---------------------------------------------------------------------------
// WMMA fragment load: 16 halves per lane = two 16B loads (works for global+LDS)
// Layout per CDNA5 ISA 7.12.2 (16-bit A 16x32): lane<16 -> K {0..7,16..23},
// lane>=16 -> K {8..15,24..31}; caller passes p = rowBase + (lane>>4)*8 + kb.
// ---------------------------------------------------------------------------
__device__ __forceinline__ v16h frag16(const _Float16* p) {
    v8h lo = *(const v8h*)(p);
    v8h hi = *(const v8h*)(p + 16);
    return __builtin_shufflevector(lo, hi, 0,1,2,3,4,5,6,7,8,9,10,11,12,13,14,15);
}

__device__ __forceinline__ float sigf(float x) { return 1.f / (1.f + __expf(-x)); }

// ---------------------------------------------------------------------------
// Generic WMMA GEMM: C[M,N] = A[M,K] * W[N,K]^T + bias, optional activation.
// A,W f16; bias f32; out f32 and/or f16. Block: 256 thr = 8 waves, each wave
// one 16-row M tile x up to 4 N tiles (block tile 128 x 64).
// Requirements: M % 128 == 0, K % 32 == 0, N % 16 == 0.
// ACT: 0 = none, 1 = relu, 2 = prelu(alpha[0])
// ---------------------------------------------------------------------------
template<int ACT>
__global__ __launch_bounds__(256) void k_gemm(
    const _Float16* __restrict__ A, const _Float16* __restrict__ W,
    const float* __restrict__ bias, const float* __restrict__ alpha,
    float* __restrict__ Cf, _Float16* __restrict__ Ch,
    int M, int N, int K)
{
    const int lane = threadIdx.x & 31;
    const int wv   = threadIdx.x >> 5;
    const int l16  = lane & 15;
    const int hf   = lane >> 4;
    const int m0   = blockIdx.x * 128 + wv * 16;
    const int n0b  = blockIdx.y * 64;
    const float aslope = (ACT == 2) ? alpha[0] : 0.f;

    v8f acc[4];
    int nts = 0;
    for (int nt = 0; nt < 4; ++nt) {
        int n0 = n0b + nt * 16;
        if (n0 >= N) break;
        ++nts;
        float bv = bias ? bias[n0 + l16] : 0.f;
        for (int r = 0; r < 8; ++r) acc[nt][r] = bv;
    }

    const _Float16* arow = A + (size_t)(m0 + l16) * K + hf * 8;
    for (int kb = 0; kb < K; kb += 32) {
        v16h af = frag16(arow + kb);
        if (kb + 32 < K) __builtin_prefetch(arow + kb + 32, 0, 0);
        for (int nt = 0; nt < nts; ++nt) {
            int n0 = n0b + nt * 16;
            const _Float16* wrow = W + (size_t)(n0 + l16) * K + hf * 8 + kb;
            v16h bf = frag16(wrow);
            acc[nt] = __builtin_amdgcn_wmma_f32_16x16x32_f16(
                false, af, false, bf, (short)0, acc[nt], false, false);
        }
    }

    for (int nt = 0; nt < nts; ++nt) {
        int col = n0b + nt * 16 + l16;
        for (int r = 0; r < 8; ++r) {
            float v = acc[nt][r];
            if (ACT == 1) v = v > 0.f ? v : 0.f;
            if (ACT == 2) v = v >= 0.f ? v : aslope * v;
            size_t idx = (size_t)(m0 + r + 8 * hf) * N + col;
            if (Cf) Cf[idx] = v;
            if (Ch) Ch[idx] = (_Float16)v;
        }
    }
}

// ---------------------------------------------------------------------------
// BiLSTM: recurrence is independent per batch row -> block owns 16 rows,
// h/c in LDS, 50 sequential steps, fused [x_t | h] (K=256) x Wfused[512,256]
// WMMA GEMM per step (wih and whh packed side by side). blockIdx.y = dir.
// ---------------------------------------------------------------------------
__global__ __launch_bounds__(128) void k_lstm(
    const _Float16* __restrict__ X,            // tr f16 [B*T, 128]
    const _Float16* __restrict__ WfF, const _Float16* __restrict__ WfB,  // [512,256]
    const float* __restrict__ biasF, const float* __restrict__ biasB,    // [512] (bih+bhh)
    float* __restrict__ outF, float* __restrict__ outB)
{
    __shared__ __align__(16) _Float16 hbuf[16 * 128];
    __shared__ float cbuf[16 * 128];
    __shared__ float gates[16 * 512];

    const int dir = blockIdx.y;
    const _Float16* W   = dir ? WfB  : WfF;
    const float* bias   = dir ? biasB : biasF;
    float* out          = dir ? outB : outF;

    const int lane = threadIdx.x & 31;
    const int wv   = threadIdx.x >> 5;     // 4 waves; wave wv -> gate cols [wv*128, +128)
    const int l16  = lane & 15;
    const int hf   = lane >> 4;
    const int m0   = blockIdx.x * 16;      // 16 batch rows

    for (int i = threadIdx.x; i < 16 * 128; i += 128) {
        hbuf[i] = (_Float16)0.f;
        cbuf[i] = 0.f;
    }
    __syncthreads();

    for (int s = 0; s < Tn; ++s) {
        const int t = dir ? (Tn - 1 - s) : s;

        v8f acc[8];
        for (int nt = 0; nt < 8; ++nt) {
            float bv = bias[wv * 128 + nt * 16 + l16];
            for (int r = 0; r < 8; ++r) acc[nt][r] = bv;
        }
        const _Float16* xrow = X + ((size_t)(m0 + l16) * Tn + t) * 128 + hf * 8;
        const _Float16* hrow = hbuf + l16 * 128 + hf * 8;
        for (int kb = 0; kb < 256; kb += 32) {
            v16h af = (kb < 128) ? frag16(xrow + kb) : frag16(hrow + (kb - 128));
            for (int nt = 0; nt < 8; ++nt) {
                const _Float16* wrow = W + (size_t)(wv * 128 + nt * 16 + l16) * 256 + hf * 8 + kb;
                v16h bf = frag16(wrow);
                acc[nt] = __builtin_amdgcn_wmma_f32_16x16x32_f16(
                    false, af, false, bf, (short)0, acc[nt], false, false);
            }
        }
        for (int nt = 0; nt < 8; ++nt)
            for (int r = 0; r < 8; ++r)
                gates[(r + 8 * hf) * 512 + wv * 128 + nt * 16 + l16] = acc[nt][r];
        __syncthreads();

        // elementwise LSTM cell update (gate order i,f,g,o)
        for (int i = threadIdx.x; i < 16 * 128; i += 128) {
            int row = i >> 7, j = i & 127;
            float gi = gates[row * 512 +       j];
            float gf = gates[row * 512 + 128 + j];
            float gg = gates[row * 512 + 256 + j];
            float go = gates[row * 512 + 384 + j];
            float c  = sigf(gf) * cbuf[i] + sigf(gi) * tanhf(gg);
            float h  = sigf(go) * tanhf(c);
            cbuf[i] = c;
            hbuf[i] = (_Float16)h;
            out[((size_t)(m0 + row) * Tn + t) * 128 + j] = h;
        }
        __syncthreads();
    }
}

// ---------------------------------------------------------------------------
// Multi-head self-attention (NH=4, DH=32) over QKV f32 [B*T, 384].
// One 64-thread block per (b, head).
// ---------------------------------------------------------------------------
__global__ __launch_bounds__(64) void k_attn(
    const float* __restrict__ QKV, const int* __restrict__ len,
    float* __restrict__ Out)
{
    const int b = blockIdx.x >> 2;
    const int h = blockIdx.x & 3;
    __shared__ float Ks[Tn * 32], Vs[Tn * 32], qrow[32], sc[Tn];
    __shared__ float invsum;
    const int L = len[b];
    const int tid = threadIdx.x;

    for (int i = tid; i < Tn * 32; i += 64) {
        int t = i >> 5, d = i & 31;
        size_t rb = ((size_t)b * Tn + t) * 384;
        Ks[i] = QKV[rb + 128 + h * 32 + d];
        Vs[i] = QKV[rb + 256 + h * 32 + d];
    }
    __syncthreads();

    for (int q = 0; q < Tn; ++q) {
        if (tid < 32) qrow[tid] = QKV[((size_t)b * Tn + q) * 384 + h * 32 + tid];
        __syncthreads();
        if (tid < Tn) {
            float s = 0.f;
            for (int d = 0; d < 32; ++d) s += qrow[d] * Ks[tid * 32 + d];
            sc[tid] = (tid < L) ? s * 0.17677669529663689f : NEG_INF; // 1/sqrt(32)
        }
        __syncthreads();
        if (tid == 0) {
            float mx = NEG_INF;
            for (int t = 0; t < Tn; ++t) mx = fmaxf(mx, sc[t]);
            float sum = 0.f;
            for (int t = 0; t < Tn; ++t) { float e = __expf(sc[t] - mx); sc[t] = e; sum += e; }
            invsum = 1.f / sum;
        }
        __syncthreads();
        if (tid < 32) {
            float o = 0.f;
            for (int t = 0; t < Tn; ++t) o += sc[t] * Vs[t * 32 + tid];
            Out[((size_t)b * Tn + q) * 128 + h * 32 + tid] = o * invsum;
        }
        __syncthreads();
    }
}

// Residual + LayerNorm over rows of 128; writes f32 (+ optional f16 copy)
__global__ __launch_bounds__(128) void k_resln(
    const float* __restrict__ A, const float* __restrict__ R,
    const float* __restrict__ g, const float* __restrict__ bt,
    float* __restrict__ of, _Float16* __restrict__ oh)
{
    const size_t row = blockIdx.x;
    const int j = threadIdx.x;
    __shared__ float red[128];
    __shared__ float st[2];
    float x = A[row * 128 + j] + R[row * 128 + j];
    red[j] = x; __syncthreads();
    for (int s = 64; s > 0; s >>= 1) { if (j < s) red[j] += red[j + s]; __syncthreads(); }
    if (j == 0) st[0] = red[0] * (1.f / 128.f);
    __syncthreads();
    float d = x - st[0];
    red[j] = d * d; __syncthreads();
    for (int s = 64; s > 0; s >>= 1) { if (j < s) red[j] += red[j + s]; __syncthreads(); }
    if (j == 0) st[1] = 1.f / sqrtf(red[0] * (1.f / 128.f) + 1e-5f);
    __syncthreads();
    float y = d * st[1] * g[j] + bt[j];
    of[row * 128 + j] = y;
    if (oh) oh[row * 128 + j] = (_Float16)y;
}

// Build attention-pool input [q, k, q-k, q*k] -> f16 [B*T, 512]
__global__ __launch_bounds__(128) void k_attin(
    const float* __restrict__ q, const float* __restrict__ keys,
    _Float16* __restrict__ dst)
{
    const size_t row = blockIdx.x;
    const int b = (int)(row / Tn);
    const int j = threadIdx.x;
    float qq = q[(size_t)b * 128 + j];
    float kk = keys[row * 128 + j];
    _Float16* d = dst + row * 512;
    d[j]        = (_Float16)qq;
    d[128 + j]  = (_Float16)kk;
    d[256 + j]  = (_Float16)(qq - kk);
    d[384 + j]  = (_Float16)(qq * kk);
}

// Small dense dot: out[m] = dot(A[m, 0:K], w) + bias[0]
__global__ void k_dot(const _Float16* __restrict__ A, const float* __restrict__ w,
                      const float* __restrict__ bias, float* __restrict__ out,
                      int M, int K)
{
    int m = blockIdx.x * blockDim.x + threadIdx.x;
    if (m >= M) return;
    float s = bias[0];
    for (int k = 0; k < K; ++k) s += (float)A[(size_t)m * K + k] * w[k];
    out[m] = s;
}

// Masked softmax over T + weighted sum of keys -> pooled [B,128]
__global__ __launch_bounds__(128) void k_poolsm(
    const float* __restrict__ score, const float* __restrict__ keys,
    const int* __restrict__ len, float* __restrict__ pooled)
{
    const int b = blockIdx.x;
    const int j = threadIdx.x;
    __shared__ float p[Tn];
    __shared__ float inv_;
    if (j < Tn) p[j] = (j < len[b]) ? score[(size_t)b * Tn + j] : NEG_INF;
    __syncthreads();
    if (j == 0) {
        float mx = NEG_INF;
        for (int t = 0; t < Tn; ++t) mx = fmaxf(mx, p[t]);
        float sum = 0.f;
        for (int t = 0; t < Tn; ++t) { float e = __expf(p[t] - mx); p[t] = e; sum += e; }
        inv_ = 1.f / sum;
    }
    __syncthreads();
    float acc = 0.f;
    for (int t = 0; t < Tn; ++t) acc += p[t] * keys[((size_t)b * Tn + t) * 128 + j];
    pooled[(size_t)b * 128 + j] = acc * inv_;
}

// ---- small utility kernels -------------------------------------------------
__global__ void k_f2h(const float* __restrict__ s, _Float16* __restrict__ d, size_t n) {
    size_t i = (size_t)blockIdx.x * blockDim.x + threadIdx.x;
    if (i < n) d[i] = (_Float16)s[i];
}
// transpose-convert [K,N] f32 -> [Npad,K] f16 (zero pad rows n >= N)
__global__ void k_pack_t(const float* __restrict__ src, _Float16* __restrict__ dst,
                         int K, int N, int Npad) {
    int idx = blockIdx.x * 256 + threadIdx.x;
    if (idx >= Npad * K) return;
    int n = idx / K, k = idx - n * K;
    dst[idx] = (n < N) ? (_Float16)src[(size_t)k * N + n] : (_Float16)0.f;
}
// convert [N,K] f32 -> f16 columns [colOff, colOff+K) of dst [N, ldDst]
__global__ void k_pack_nt(const float* __restrict__ src, _Float16* __restrict__ dst,
                          int N, int K, int ldDst, int colOff) {
    int idx = blockIdx.x * 256 + threadIdx.x;
    if (idx >= N * K) return;
    int n = idx / K, k = idx - n * K;
    dst[(size_t)n * ldDst + colOff + k] = (_Float16)src[idx];
}
__global__ void k_addv(const float* a, const float* b, float* d, int n) {
    int i = blockIdx.x * blockDim.x + threadIdx.x;
    if (i < n) d[i] = a[i] + b[i];
}
__global__ void k_avg2(const float* a, const float* b, float* d, size_t n) {
    size_t i = (size_t)blockIdx.x * blockDim.x + threadIdx.x;
    if (i < n) d[i] = 0.5f * (a[i] + b[i]);
}
// x2[b] = [profile(64) | pooled1(128) | pooled2(128)] -> f16 [B,320]
__global__ __launch_bounds__(128) void k_concatx(
    const float* __restrict__ prof, const float* __restrict__ p1,
    const float* __restrict__ p2, _Float16* __restrict__ x2)
{
    const int b = blockIdx.x, j = threadIdx.x;
    _Float16* xr = x2 + (size_t)b * 320;
    if (j < 64) xr[j] = (_Float16)prof[(size_t)b * 64 + j];
    xr[64 + j]  = (_Float16)p1[(size_t)b * 128 + j];
    xr[192 + j] = (_Float16)p2[(size_t)b * 128 + j];
}

// ===========================================================================
extern "C" void kernel_launch(void* const* d_in, const int* in_sizes, int n_in,
                              void* d_out, int out_size, void* d_ws, size_t ws_size,
                              hipStream_t stream) {
    (void)in_sizes; (void)n_in; (void)out_size; (void)ws_size;

    const float* query   = (const float*)d_in[0];
    const float* keys    = (const float*)d_in[1];
    const float* profile = (const float*)d_in[2];
    const int*   klen    = (const int*)d_in[3];
    const float* wq = (const float*)d_in[4];  const float* bq = (const float*)d_in[5];
    const float* wk = (const float*)d_in[6];  const float* bk = (const float*)d_in[7];
    const float* wv = (const float*)d_in[8];  const float* bv = (const float*)d_in[9];
    const float* f1w = (const float*)d_in[10]; const float* f1b = (const float*)d_in[11];
    const float* f2w = (const float*)d_in[12]; const float* f2b = (const float*)d_in[13];
    const float* ln_g = (const float*)d_in[14]; const float* ln_b = (const float*)d_in[15];
    // pools p1 = inputs 16..23, p2 = 24..31
    const float* pw1[2]  = { (const float*)d_in[16], (const float*)d_in[24] };
    const float* pb1[2]  = { (const float*)d_in[17], (const float*)d_in[25] };
    const float* pa1[2]  = { (const float*)d_in[18], (const float*)d_in[26] };
    const float* pw2[2]  = { (const float*)d_in[19], (const float*)d_in[27] };
    const float* pb2[2]  = { (const float*)d_in[20], (const float*)d_in[28] };
    const float* pa2[2]  = { (const float*)d_in[21], (const float*)d_in[29] };
    const float* pfcw[2] = { (const float*)d_in[22], (const float*)d_in[30] };
    const float* pfcb[2] = { (const float*)d_in[23], (const float*)d_in[31] };
    const float* fw_wih = (const float*)d_in[32]; const float* fw_whh = (const float*)d_in[33];
    const float* fw_bih = (const float*)d_in[34]; const float* fw_bhh = (const float*)d_in[35];
    const float* bw_wih = (const float*)d_in[36]; const float* bw_whh = (const float*)d_in[37];
    const float* bw_bih = (const float*)d_in[38]; const float* bw_bhh = (const float*)d_in[39];
    const float* d1_w = (const float*)d_in[40]; const float* d1_b = (const float*)d_in[41];
    const float* d1_a = (const float*)d_in[42];
    const float* d2_w = (const float*)d_in[43]; const float* d2_b = (const float*)d_in[44];
    const float* d2_a = (const float*)d_in[45];
    const float* d3_w = (const float*)d_in[46]; const float* d3_b = (const float*)d_in[47];

    char* ws = (char*)d_ws;
    const size_t MB = 1048576ULL;
    size_t cur = 0;
    auto alloc = [&](size_t bytes) { size_t r = cur; cur += (bytes + 255) & ~(size_t)255; return r; };

    // ---- small persistent arena (packed weights + tiny activations) -------
    size_t o_wqkv  = alloc(384 * 128 * 2);
    size_t o_bqkv  = alloc(384 * 4);
    size_t o_f1wT  = alloc(512 * 128 * 2);
    size_t o_f2wT  = alloc(128 * 512 * 2);
    size_t o_pw1T[2] = { alloc(32 * 512 * 2), alloc(32 * 512 * 2) };
    size_t o_pw2T[2] = { alloc(16 * 32 * 2),  alloc(16 * 32 * 2) };
    size_t o_d1wT  = alloc(64 * 320 * 2);
    size_t o_d2wT  = alloc(32 * 64 * 2);
    size_t o_wfF   = alloc(512 * 256 * 2);
    size_t o_wfB   = alloc(512 * 256 * 2);
    size_t o_bF    = alloc(512 * 4);
    size_t o_bB    = alloc(512 * 4);
    size_t o_pooled[2] = { alloc((size_t)Bn * 128 * 4), alloc((size_t)Bn * 128 * 4) };
    size_t o_score = alloc((size_t)Mn * 4);
    size_t o_x2    = alloc((size_t)Bn * 320 * 2);
    size_t o_dh1   = alloc((size_t)Bn * 64 * 2);
    size_t o_dh2   = alloc((size_t)Bn * 32 * 2);

    // ---- big overlaid regions ---------------------------------------------
    size_t BIG = (cur + MB - 1) & ~(MB - 1);
    size_t RA = BIG;             // 210 MB
    size_t RB = RA + 210 * MB;   // 315 MB
    size_t RC = RB + 315 * MB;   // 220 MB
    size_t RD = RC + 220 * MB;   // 170 MB

    size_t o_keysh  = RA;               // f16 [M,128]
    size_t o_qkv    = RB;               // f32 [M,384]
    size_t o_attn   = RC;               // f32 [M,128]
    size_t o_tr1f   = RC + 110 * MB;    // f32 [M,128]
    size_t o_tr1h   = RD;               // f16 [M,128]
    size_t o_ffh    = RA;               // f16 [M,512]  (keysh dead)
    size_t o_ff2    = RD + 55 * MB;     // f32 [M,128]
    size_t o_trf    = RB;               // f32 [M,128]  (qkv dead)
    size_t o_trh    = RB + 110 * MB;    // f16 [M,128]
    size_t o_attin  = RA;               // f16 [M,512]  (ffh dead)
    size_t o_h1[2]  = { RC, RD + 110 * MB };         // f16 [M,32]
    size_t o_h2[2]  = { RC + 16 * MB, RD + 130 * MB };// f16 [M,16]
    size_t o_outfw  = RC;               // f32 [M,128]  (after pool1)
    size_t o_outbw  = RC + 110 * MB;    // f32 [M,128]
    size_t o_lstm   = RD;               // f32 [M,128]  (tr1h/ff2 dead)

    auto H  = [&](size_t o) { return (_Float16*)(ws + o); };
    auto F  = [&](size_t o) { return (float*)(ws + o); };

    // ======================= weight packing =================================
    {
        size_t n = (size_t)Mn * 128;
        k_f2h<<<(unsigned)((n + 255) / 256), 256, 0, stream>>>(keys, H(o_keysh), n);
    }
    k_pack_t<<<(128*128 + 255)/256, 256, 0, stream>>>(wq, H(o_wqkv),             128, 128, 128);
    k_pack_t<<<(128*128 + 255)/256, 256, 0, stream>>>(wk, H(o_wqkv) + 128 * 128, 128, 128, 128);
    k_pack_t<<<(128*128 + 255)/256, 256, 0, stream>>>(wv, H(o_wqkv) + 256 * 128, 128, 128, 128);
    hipMemcpyAsync(F(o_bqkv),       bq, 128 * 4, hipMemcpyDeviceToDevice, stream);
    hipMemcpyAsync(F(o_bqkv) + 128, bk, 128 * 4, hipMemcpyDeviceToDevice, stream);
    hipMemcpyAsync(F(o_bqkv) + 256, bv, 128 * 4, hipMemcpyDeviceToDevice, stream);
    k_pack_t<<<(512*128 + 255)/256, 256, 0, stream>>>(f1w, H(o_f1wT), 128, 512, 512);
    k_pack_t<<<(128*512 + 255)/256, 256, 0, stream>>>(f2w, H(o_f2wT), 512, 128, 128);
    for (int p = 0; p < 2; ++p) {
        k_pack_t<<<(32*512 + 255)/256, 256, 0, stream>>>(pw1[p], H(o_pw1T[p]), 512, 32, 32);
        k_pack_t<<<(16*32  + 255)/256, 256, 0, stream>>>(pw2[p], H(o_pw2T[p]), 32, 16, 16);
    }
    k_pack_t<<<(64*320 + 255)/256, 256, 0, stream>>>(d1_w, H(o_d1wT), 320, 64, 64);
    k_pack_t<<<(32*64  + 255)/256, 256, 0, stream>>>(d2_w, H(o_d2wT), 64, 32, 32);
    k_pack_nt<<<(512*128 + 255)/256, 256, 0, stream>>>(fw_wih, H(o_wfF), 512, 128, 256, 0);
    k_pack_nt<<<(512*128 + 255)/256, 256, 0, stream>>>(fw_whh, H(o_wfF), 512, 128, 256, 128);
    k_pack_nt<<<(512*128 + 255)/256, 256, 0, stream>>>(bw_wih, H(o_wfB), 512, 128, 256, 0);
    k_pack_nt<<<(512*128 + 255)/256, 256, 0, stream>>>(bw_whh, H(o_wfB), 512, 128, 256, 128);
    k_addv<<<2, 256, 0, stream>>>(fw_bih, fw_bhh, F(o_bF), 512);
    k_addv<<<2, 256, 0, stream>>>(bw_bih, bw_bhh, F(o_bB), 512);

    const int MB128 = Mn / 128;  // 1600

    // ======================= transformer ====================================
    k_gemm<0><<<dim3(MB128, 6), 256, 0, stream>>>(H(o_keysh), H(o_wqkv), F(o_bqkv),
        nullptr, F(o_qkv), nullptr, Mn, 384, 128);
    k_attn<<<Bn * 4, 64, 0, stream>>>(F(o_qkv), klen, F(o_attn));
    k_resln<<<Mn, 128, 0, stream>>>(F(o_attn), keys, ln_g, ln_b, F(o_tr1f), H(o_tr1h));
    k_gemm<1><<<dim3(MB128, 8), 256, 0, stream>>>(H(o_tr1h), H(o_f1wT), f1b,
        nullptr, nullptr, H(o_ffh), Mn, 512, 128);
    k_gemm<0><<<dim3(MB128, 2), 256, 0, stream>>>(H(o_ffh), H(o_f2wT), f2b,
        nullptr, F(o_ff2), nullptr, Mn, 128, 512);
    k_resln<<<Mn, 128, 0, stream>>>(F(o_ff2), F(o_tr1f), ln_g, ln_b, F(o_trf), H(o_trh));

    // ======================= pool 1 =========================================
    k_attin<<<Mn, 128, 0, stream>>>(query, F(o_trf), H(o_attin));
    k_gemm<2><<<dim3(MB128, 1), 256, 0, stream>>>(H(o_attin), H(o_pw1T[0]), pb1[0],
        pa1[0], nullptr, H(o_h1[0]), Mn, 32, 512);
    k_gemm<2><<<dim3(MB128, 1), 256, 0, stream>>>(H(o_h1[0]), H(o_pw2T[0]), pb2[0],
        pa2[0], nullptr, H(o_h2[0]), Mn, 16, 32);
    k_dot<<<(Mn + 255)/256, 256, 0, stream>>>(H(o_h2[0]), pfcw[0], pfcb[0], F(o_score), Mn, 16);
    k_poolsm<<<Bn, 128, 0, stream>>>(F(o_score), F(o_trf), klen, F(o_pooled[0]));

    // ======================= BiLSTM =========================================
    k_lstm<<<dim3(Bn / 16, 2), 128, 0, stream>>>(H(o_trh), H(o_wfF), H(o_wfB),
        F(o_bF), F(o_bB), F(o_outfw), F(o_outbw));
    {
        size_t n = (size_t)Mn * 128;
        k_avg2<<<(unsigned)((n + 255) / 256), 256, 0, stream>>>(F(o_outfw), F(o_outbw), F(o_lstm), n);
    }

    // ======================= pool 2 =========================================
    k_attin<<<Mn, 128, 0, stream>>>(query, F(o_lstm), H(o_attin));
    k_gemm<2><<<dim3(MB128, 1), 256, 0, stream>>>(H(o_attin), H(o_pw1T[1]), pb1[1],
        pa1[1], nullptr, H(o_h1[1]), Mn, 32, 512);
    k_gemm<2><<<dim3(MB128, 1), 256, 0, stream>>>(H(o_h1[1]), H(o_pw2T[1]), pb2[1],
        pa2[1], nullptr, H(o_h2[1]), Mn, 16, 32);
    k_dot<<<(Mn + 255)/256, 256, 0, stream>>>(H(o_h2[1]), pfcw[1], pfcb[1], F(o_score), Mn, 16);
    k_poolsm<<<Bn, 128, 0, stream>>>(F(o_score), F(o_lstm), klen, F(o_pooled[1]));

    // ======================= final DNN ======================================
    k_concatx<<<Bn, 128, 0, stream>>>(profile, F(o_pooled[0]), F(o_pooled[1]), H(o_x2));
    k_gemm<2><<<dim3(Bn / 128, 1), 256, 0, stream>>>(H(o_x2), H(o_d1wT), d1_b,
        d1_a, nullptr, H(o_dh1), Bn, 64, 320);
    k_gemm<2><<<dim3(Bn / 128, 1), 256, 0, stream>>>(H(o_dh1), H(o_d2wT), d2_b,
        d2_a, nullptr, H(o_dh2), Bn, 32, 64);
    k_dot<<<(Bn + 255)/256, 256, 0, stream>>>(H(o_dh2), d3_w, d3_b, (float*)d_out, Bn, 32);
}
